// SiamPointNet_28200755266173
// MI455X (gfx1250) — matrix operand order repaired and checked by reading, compiled, not run on hardware
//
#include <hip/hip_runtime.h>
#include <math.h>

// ---------------------------------------------------------------------------
// Types
// ---------------------------------------------------------------------------
typedef _Float16 f16;
typedef __attribute__((ext_vector_type(8)))  f16   h8;
typedef __attribute__((ext_vector_type(16))) f16   v16h;
typedef __attribute__((ext_vector_type(8)))  float v8f;

#define EPSB 1e-5f
static constexpr int NB   = 4;
static constexpr int NPTS = 8192;
static constexpr int KNN  = 16;

// ---------------------------------------------------------------------------
// Helpers
// ---------------------------------------------------------------------------
__device__ inline void atomicMaxFloat(float* addr, float val) {
    // valid when addr initialized to -inf; monotone int/uint trick
    if (val >= 0.0f) atomicMax((int*)addr, __float_as_int(val));
    else             atomicMin((unsigned int*)addr, __float_as_uint(val));
}

__device__ inline v8f wmma_f16(v16h a, v16h b, v8f c) {
#if __HIP_DEVICE_COMPILE__
    return __builtin_amdgcn_wmma_f32_16x16x32_f16(
        /*neg_a=*/false, a, /*neg_b=*/false, b,
        /*c_mod=*/(short)0, c, /*reuse_a=*/false, /*reuse_b=*/false);
#else
    (void)a; (void)b;
    return c;
#endif
}

__global__ void fill_f32(float* __restrict__ p, float v, int total) {
    int i = blockIdx.x * blockDim.x + threadIdx.x;
    if (i < total) p[i] = v;
}

// ---------------------------------------------------------------------------
// Weight prep: fold BN into f16 weights (+ f32 combined bias), pad Cin->Cinp
//   y = scale*(W x + b) + shift  ==>  W' = scale*W (f16),  b' = scale*b + shift
// ---------------------------------------------------------------------------
__global__ void prep_weights(const float* __restrict__ w, int w_row_stride, int w_col_off,
                             const float* __restrict__ bias,
                             const float* __restrict__ bng, const float* __restrict__ bnb,
                             const float* __restrict__ bnm, const float* __restrict__ bnv,
                             f16* __restrict__ wh, float* __restrict__ bias2,
                             int cin, int cinp, int cout) {
    int i = blockIdx.x * blockDim.x + threadIdx.x;
    if (i >= cout * cinp) return;
    int o = i / cinp, k = i % cinp;
    float scale = 1.0f, shift = 0.0f;
    if (bng) {
        scale = bng[o] * rsqrtf(bnv[o] + EPSB);
        shift = bnb[o] - bnm[o] * scale;
    }
    float wv = (k < cin) ? w[(size_t)o * w_row_stride + w_col_off + k] * scale : 0.0f;
    wh[i] = (f16)wv;
    if (k == 0 && bias2) bias2[o] = (bias ? bias[o] * scale : 0.0f) + shift;
}

// per-batch feature-transform weights: Wh[b][e][c] = T[b][c][e] (f32 -> f16)
__global__ void prep_transW(const float* __restrict__ T, f16* __restrict__ wh,
                            int k, int total) {
    int i = blockIdx.x * blockDim.x + threadIdx.x;
    if (i >= total) return;
    int b = i / (k * k);
    int r = (i / k) % k;
    int c = i % k;
    wh[i] = (f16)T[(size_t)b * k * k + (size_t)c * k + r];
}

// ---------------------------------------------------------------------------
// Core fused conv1x1 (+BN folded) (+ReLU) (+global max-pool) via WMMA f16
//   X: [B][Cin][N] (TIN), Y: [B][Cout][N] (TOUT)
//   W: f16 [Cout][Cinp] (optionally per-batch), bias: f32 (optionally per-batch)
//   Block: 256 thr = 8 waves -> 64(cout) x 64(points) tile, K-loop by 64.
//   Wave (mw,nw): 16(cout) x 32(points) -> 2 accumulators, 4 WMMA per K stage.
// ---------------------------------------------------------------------------
template <typename TIN, typename TOUT, bool RELU, bool REDUCE_MAX>
__global__ __launch_bounds__(256)
void conv1x1_wmma(const TIN* __restrict__ X, TOUT* __restrict__ Y,
                  const f16* __restrict__ W, const float* __restrict__ bias,
                  float* __restrict__ gmax,
                  int cin, int cinp, int cout, int npts,
                  int w_batch_stride, int bias_batch_stride) {
    constexpr int KP = 72;                      // padded row: 144B, 16B-aligned, 36-word stride
    __shared__ __align__(16) f16 Xs[64][KP];    // [point][k]
    __shared__ __align__(16) f16 Ws[64][KP];    // [m][k]

    const int b    = blockIdx.z;
    const int n0   = blockIdx.x * 64;
    const int m0   = blockIdx.y * 64;
    const int tid  = threadIdx.x;
    const int lane = tid & 31;
    const int wave = tid >> 5;
    const int mw   = wave & 3;   // 4 m-subtiles of 16
    const int nw   = wave >> 2;  // 2 n-supertiles of 32

    const f16*   Wb    = W + (size_t)b * w_batch_stride;
    const float* biasb = bias ? (bias + (size_t)b * bias_batch_stride) : nullptr;

    v8f acc0 = {}, acc1 = {};

    // staging assignments (fixed per thread)
    const int xkk = tid >> 2;          // 0..63 : k column staged by this thread
    const int xnn = (tid & 3) * 16;    // point start (16 consecutive points)
    const int wm  = tid >> 2;          // 0..63 : weight row
    const int wkk = (tid & 3) * 16;    // k start (16 consecutive halfs)

    for (int k0 = 0; k0 < cinp; k0 += 64) {
        // ---- stage X tile (64k x 64n), store transposed [n][k] ----
        {
            int c = k0 + xkk;
            f16 vals[16];
            if (c < cin) {
                const TIN* src = X + (((size_t)b * cin + c) * npts + n0 + xnn);
#pragma unroll
                for (int j = 0; j < 16; ++j) vals[j] = (f16)(float)src[j];
                if (c + 64 < cin) __builtin_prefetch(src + (size_t)64 * npts, 0, 1);
            } else {
#pragma unroll
                for (int j = 0; j < 16; ++j) vals[j] = (f16)0.0f;
            }
#pragma unroll
            for (int j = 0; j < 16; ++j) Xs[xnn + j][xkk] = vals[j];
        }
        // ---- stage W tile (64m x 64k): 16 contiguous halfs per thread ----
        {
            const f16* src = Wb + ((size_t)(m0 + wm) * cinp + k0 + wkk);
            *(h8*)&Ws[wm][wkk]     = *(const h8*)(src);
            *(h8*)&Ws[wm][wkk + 8] = *(const h8*)(src + 8);
        }
        __syncthreads();

        // ---- A(16x32)/B(32x16) fragments, same contiguous k order (k = 16*half + s) in A and B ----
        const int mm    = mw * 16 + (lane & 15);
        const int khalf = (lane >> 4) * 8;    // unused (fragment loads use koff below)
        const int koff  = (lane >> 4) * 16;   // lanes 16-31: K=16..31 of chunk
        const int nA    = nw * 32 + (lane & 15);
        const int nBr   = nA + 16;

        union { v16h v; h8 h[2]; } A0, A1, B00, B01, B10, B11;
        A0.h[0]  = *(const h8*)&Ws[mm][koff];
        A0.h[1]  = *(const h8*)&Ws[mm][koff + 8];
        A1.h[0]  = *(const h8*)&Ws[mm][32 + koff];
        A1.h[1]  = *(const h8*)&Ws[mm][32 + koff + 8];
        B00.h[0] = *(const h8*)&Xs[nA][koff];
        B00.h[1] = *(const h8*)&Xs[nA][koff + 8];
        B01.h[0] = *(const h8*)&Xs[nA][32 + koff];
        B01.h[1] = *(const h8*)&Xs[nA][32 + koff + 8];
        B10.h[0] = *(const h8*)&Xs[nBr][koff];
        B10.h[1] = *(const h8*)&Xs[nBr][koff + 8];
        B11.h[0] = *(const h8*)&Xs[nBr][32 + koff];
        B11.h[1] = *(const h8*)&Xs[nBr][32 + koff + 8];

        acc0 = wmma_f16(A0.v, B00.v, acc0);
        acc0 = wmma_f16(A1.v, B01.v, acc0);
        acc1 = wmma_f16(A0.v, B10.v, acc1);
        acc1 = wmma_f16(A1.v, B11.v, acc1);

        __syncthreads();
    }

    // ---- epilogue: C/D layout -> lane(n), vgpr(m) ----
    const int nn    = lane & 15;
    const int mhalf = (lane >> 4) * 8;   // lanes 16-31 hold M=8..15
#pragma unroll
    for (int v = 0; v < 8; ++v) {
        int   m   = m0 + mw * 16 + mhalf + v;
        float bia = biasb ? biasb[m] : 0.0f;
        float v0  = acc0[v] + bia;
        float v1  = acc1[v] + bia;
        if (RELU) { v0 = fmaxf(v0, 0.0f); v1 = fmaxf(v1, 0.0f); }
        if (REDUCE_MAX) {
            float r = fmaxf(v0, v1);
#pragma unroll
            for (int s = 1; s < 16; s <<= 1)
                r = fmaxf(r, __shfl_xor(r, s, 32));
            if (nn == 0) atomicMaxFloat(&gmax[(size_t)b * cout + m], r);
        } else {
            size_t base = ((size_t)b * cout + m) * npts + n0 + nw * 32 + nn;
            Y[base]      = (TOUT)v0;
            Y[base + 16] = (TOUT)v1;
        }
    }
}

// ---------------------------------------------------------------------------
// Small dense (FC) on pooled vectors: y[b][o] = act(bn(W x + b)) (+identity)
// ---------------------------------------------------------------------------
__global__ void dense_kernel(const float* __restrict__ x,
                             const float* __restrict__ w, int w_row_stride,
                             const float* __restrict__ bias,
                             const float* __restrict__ bng, const float* __restrict__ bnb,
                             const float* __restrict__ bnm, const float* __restrict__ bnv,
                             float* __restrict__ y, int cin, int cout,
                             int relu, int ident_k, int total) {
    int i = blockIdx.x * blockDim.x + threadIdx.x;
    if (i >= total) return;
    int b = i / cout, o = i % cout;
    const float* xb = x + (size_t)b * cin;
    const float* wr = w + (size_t)o * w_row_stride;
    float s = 0.0f;
    for (int c = 0; c < cin; ++c) s += wr[c] * xb[c];
    if (bias) s += bias[o];
    if (bng) {
        float sc = bng[o] * rsqrtf(bnv[o] + EPSB);
        s = sc * s + (bnb[o] - bnm[o] * sc);
    }
    if (relu) s = fmaxf(s, 0.0f);
    if (ident_k) s += ((o / ident_k) == (o % ident_k)) ? 1.0f : 0.0f;
    y[i] = s;
}

// ---------------------------------------------------------------------------
// [B,N,3] -> [B,3,N]
// ---------------------------------------------------------------------------
__global__ void transpose_in(const float* __restrict__ ep, float* __restrict__ xT,
                             int npts, int total) {
    int i = blockIdx.x * blockDim.x + threadIdx.x;
    if (i >= total) return;
    int b = i / npts, n = i % npts;
    const float* s = ep + ((size_t)b * npts + n) * 3;
    float* d = xT + (size_t)b * 3 * npts + n;
    d[0] = s[0]; d[(size_t)npts] = s[1]; d[2 * (size_t)npts] = s[2];
}

// xe[b][e][n] = sum_c xT[b][c][n] * T[b][c*3+e]
__global__ void apply_trans3(const float* __restrict__ xT, const float* __restrict__ T,
                             float* __restrict__ xe, int npts, int total) {
    int i = blockIdx.x * blockDim.x + threadIdx.x;
    if (i >= total) return;
    int b = i / npts, n = i % npts;
    const float* xb = xT + (size_t)b * 3 * npts + n;
    float x0 = xb[0], x1 = xb[(size_t)npts], x2 = xb[2 * (size_t)npts];
    const float* t = T + (size_t)b * 9;
    float* d = xe + (size_t)b * 3 * npts + n;
#pragma unroll
    for (int e = 0; e < 3; ++e)
        d[(size_t)e * npts] = x0 * t[e] + x1 * t[3 + e] + x2 * t[6 + e];
}

// ---------------------------------------------------------------------------
// nearest feature difference: f[b][c][n] = mean_k |raw - query[.., idx[b][n][k]]|
// ---------------------------------------------------------------------------
__global__ void nfd_kernel(const float* __restrict__ raw, const float* __restrict__ query,
                           const int* __restrict__ idx, float* __restrict__ f,
                           int npts, int total) {
    int i = blockIdx.x * blockDim.x + threadIdx.x;
    if (i >= total) return;
    int n = i % npts;
    int c = (i / npts) & 63;
    int b = i / (npts * 64);
    float r = raw[i];
    const int*   id = idx + ((size_t)b * npts + n) * KNN;
    const float* q  = query + ((size_t)b * 64 + c) * npts;
    float s = 0.0f;
#pragma unroll
    for (int k = 0; k < KNN; ++k) s += fabsf(r - q[id[k]]);
    f[i] = s * (1.0f / KNN);
}

// ---------------------------------------------------------------------------
// Final head: mlp1(64->32)+BN+LeakyReLU(0.2) -> mlp2(32->1,no bias) -> sigmoid
// ---------------------------------------------------------------------------
__global__ __launch_bounds__(256)
void head_kernel(const float* __restrict__ f,
                 const float* __restrict__ w1, const float* __restrict__ b1,
                 const float* __restrict__ bng, const float* __restrict__ bnb,
                 const float* __restrict__ bnm, const float* __restrict__ bnv,
                 const float* __restrict__ w2, float* __restrict__ out, int npts) {
    __shared__ float Wsh[32 * 64];
    __shared__ float sc[32], sh[32], w2s[32];
    int tid = threadIdx.x;
    for (int i = tid; i < 32 * 64; i += 256) Wsh[i] = w1[i];
    if (tid < 32) {
        float s = bng[tid] * rsqrtf(bnv[tid] + EPSB);
        sc[tid]  = s;
        sh[tid]  = bnb[tid] - bnm[tid] * s + s * b1[tid];
        w2s[tid] = w2[tid];
    }
    __syncthreads();
    int p = blockIdx.x * 256 + tid;
    if (p >= NB * npts) return;
    int b = p / npts, n = p % npts;
    float h[32];
#pragma unroll
    for (int j = 0; j < 32; ++j) h[j] = 0.0f;
    const float* fb = f + (size_t)b * 64 * npts + n;
    for (int c = 0; c < 64; ++c) {
        float fv = fb[(size_t)c * npts];
#pragma unroll
        for (int j = 0; j < 32; ++j) h[j] += Wsh[j * 64 + c] * fv;
    }
    float o = 0.0f;
#pragma unroll
    for (int j = 0; j < 32; ++j) {
        float t = sc[j] * h[j] + sh[j];
        t = (t > 0.0f) ? t : 0.2f * t;
        o += w2s[j] * t;
    }
    out[p] = 1.0f / (1.0f + expf(-o));
}

// ---------------------------------------------------------------------------
// Host side: parameter tree decoding + orchestration
// ---------------------------------------------------------------------------
struct BnP  { const float *b, *g, *m, *v; };
struct LinP { const float *b, *w; };
struct StnP { BnP bn1, bn2, bn3, bn4, bn5; LinP c1, c2, c3, f1, f2, f3; };

extern "C" void kernel_launch(void* const* d_in, const int* in_sizes, int n_in,
                              void* d_out, int out_size, void* d_ws, size_t ws_size,
                              hipStream_t stream) {
    (void)in_sizes; (void)n_in; (void)out_size; (void)ws_size;
    const float* ep0   = (const float*)d_in[0];
    const float* ep1   = (const float*)d_in[1];
    const int*   idx01 = (const int*)d_in[2];
    const int*   idx10 = (const int*)d_in[3];

    // --- params flattened jax-style (dict keys sorted alphabetically) ---
    int pi = 4;
    auto nf = [&]() { return (const float*)d_in[pi++]; };
    auto rdBn  = [&](BnP& p)  { p.b = nf(); p.g = nf(); p.m = nf(); p.v = nf(); };
    auto rdLin = [&](LinP& p) { p.b = nf(); p.w = nf(); };
    auto rdStn = [&](StnP& p) {
        rdBn(p.bn1); rdBn(p.bn2); rdBn(p.bn3); rdBn(p.bn4); rdBn(p.bn5);
        rdLin(p.c1); rdLin(p.c2); rdLin(p.c3);
        rdLin(p.f1); rdLin(p.f2); rdLin(p.f3);
    };
    BnP bn_m1; rdBn(bn_m1);                 // params.bn_m1
    LinP mlp1; rdLin(mlp1);                 // params.mlp1
    const float* mlp2w = nf();              // params.mlp2_w
    BnP nbn1, nbn2, nbn3, nbn4;             // params.net
    rdBn(nbn1); rdBn(nbn2); rdBn(nbn3); rdBn(nbn4);
    LinP nc1, nc2, nc3, nc4;
    rdLin(nc1); rdLin(nc2); rdLin(nc3); rdLin(nc4);
    BnP fbn1, fbn2, fbn3;                   // params.net.feat
    rdBn(fbn1); rdBn(fbn2); rdBn(fbn3);
    LinP fc1, fc2, fc3;
    rdLin(fc1); rdLin(fc2); rdLin(fc3);
    StnP fstn; rdStn(fstn);                 // feat.fstn (sorted before feat.stn)
    StnP stn;  rdStn(stn);

    // --- workspace bump allocator ---
    char* wsb = (char*)d_ws;
    size_t off = 0;
    auto alloc = [&](size_t bytes) -> void* {
        void* p = wsb + off;
        off = (off + bytes + 255) & ~(size_t)255;
        return p;
    };
    const size_t F16BN = (size_t)NB * NPTS * 2;  // f16 bytes per channel-plane set
    const size_t F32BN = (size_t)NB * NPTS * 4;

    // prepped weights/biases (shared by both clouds); cinp padded to mult of 64
    f16*   pw_s1 = (f16*)alloc(64  * 64  * 2);  float* pb_s1 = (float*)alloc(64  * 4);
    f16*   pw_s2 = (f16*)alloc(128 * 64  * 2);  float* pb_s2 = (float*)alloc(128 * 4);
    f16*   pw_s3 = (f16*)alloc(1024* 128 * 2);  float* pb_s3 = (float*)alloc(1024* 4);
    f16*   pw_f1 = (f16*)alloc(64  * 64  * 2);  float* pb_f1 = (float*)alloc(64  * 4);
    f16*   pw_f2 = (f16*)alloc(128 * 64  * 2);  float* pb_f2 = (float*)alloc(128 * 4);
    f16*   pw_f3 = (f16*)alloc(1024* 128 * 2);  float* pb_f3 = (float*)alloc(1024* 4);
    f16*   pw_e1 = (f16*)alloc(64  * 64  * 2);  float* pb_e1 = (float*)alloc(64  * 4);
    f16*   pw_e2 = (f16*)alloc(128 * 64  * 2);  float* pb_e2 = (float*)alloc(128 * 4);
    f16*   pw_e3 = (f16*)alloc(1024* 128 * 2);  float* pb_e3 = (float*)alloc(1024* 4);
    f16*   pw_h1 = (f16*)alloc(512 * 64  * 2);
    f16*   pw_h2 = (f16*)alloc(256 * 512 * 2);  float* pb_h2 = (float*)alloc(256 * 4);
    f16*   pw_h3 = (f16*)alloc(128 * 256 * 2);  float* pb_h3 = (float*)alloc(128 * 4);
    f16*   pw_h4 = (f16*)alloc(64  * 128 * 2);  float* pb_h4 = (float*)alloc(64  * 4);

    float* outA = (float*)alloc(64 * F32BN);   // model output cloud0 [B,64,N]
    float* outB = (float*)alloc(64 * F32BN);   // model output cloud1
    float* fd0  = (float*)alloc(64 * F32BN);   // NFD(out0, out1)
    float* fd1  = (float*)alloc(64 * F32BN);   // NFD(out1, out0)

    // per-cloud scratch (reused sequentially)
    float* xT   = (float*)alloc(3 * F32BN);
    float* xe   = (float*)alloc(3 * F32BN);
    f16*   s1   = (f16*)alloc(64  * F16BN);
    f16*   s2   = (f16*)alloc(128 * F16BN);
    f16*   t1   = (f16*)alloc(64  * F16BN);
    f16*   t2   = (f16*)alloc(128 * F16BN);
    f16*   e1   = (f16*)alloc(64  * F16BN);
    f16*   pf   = (f16*)alloc(64  * F16BN);
    f16*   e2   = (f16*)alloc(128 * F16BN);
    f16*   d1   = (f16*)alloc(512 * F16BN);
    f16*   d2   = (f16*)alloc(256 * F16BN);
    f16*   d3   = (f16*)alloc(128 * F16BN);
    float* pool = (float*)alloc(NB * 1024 * 4);
    float* gmax = (float*)alloc(NB * 1024 * 4);
    float* v512 = (float*)alloc(NB * 512 * 4);
    float* v256 = (float*)alloc(NB * 256 * 4);
    float* tr3  = (float*)alloc(NB * 9 * 4);
    float* tr64 = (float*)alloc(NB * 4096 * 4);
    f16*   tw   = (f16*)alloc(NB * 4096 * 2);
    float* gb   = (float*)alloc(NB * 512 * 4);

    auto prep = [&](const LinP& l, int wstride, int coloff, const BnP* bn,
                    f16* wh, float* b2, int cin, int cinp, int cout, bool use_bias) {
        int tot = cout * cinp;
        prep_weights<<<(tot + 255) / 256, 256, 0, stream>>>(
            l.w, wstride, coloff, use_bias ? l.b : nullptr,
            bn ? bn->g : nullptr, bn ? bn->b : nullptr,
            bn ? bn->m : nullptr, bn ? bn->v : nullptr,
            wh, b2, cin, cinp, cout);
    };
    prep(stn.c1, 3,    0,    &stn.bn1,  pw_s1, pb_s1, 3,   64,  64,   true);
    prep(stn.c2, 64,   0,    &stn.bn2,  pw_s2, pb_s2, 64,  64,  128,  true);
    prep(stn.c3, 128,  0,    &stn.bn3,  pw_s3, pb_s3, 128, 128, 1024, true);
    prep(fstn.c1,64,   0,    &fstn.bn1, pw_f1, pb_f1, 64,  64,  64,   true);
    prep(fstn.c2,64,   0,    &fstn.bn2, pw_f2, pb_f2, 64,  64,  128,  true);
    prep(fstn.c3,128,  0,    &fstn.bn3, pw_f3, pb_f3, 128, 128, 1024, true);
    prep(fc1,    3,    0,    &fbn1,     pw_e1, pb_e1, 3,   64,  64,   true);
    prep(fc2,    64,   0,    &fbn2,     pw_e2, pb_e2, 64,  64,  128,  true);
    prep(fc3,    128,  0,    &fbn3,     pw_e3, pb_e3, 128, 128, 1024, true);
    prep(nc1,    1088, 1024, &nbn1,     pw_h1, nullptr, 64, 64, 512,  false);
    prep(nc2,    512,  0,    &nbn2,     pw_h2, pb_h2, 512, 512, 256,  true);
    prep(nc3,    256,  0,    &nbn3,     pw_h3, pb_h3, 256, 256, 128,  true);
    prep(nc4,    128,  0,    &nbn4,     pw_h4, pb_h4, 128, 128, 64,   true);

    auto dense = [&](const float* x, const float* w, int wstride, const float* bias,
                     const BnP* bn, float* y, int cin, int cout, int relu, int identk) {
        int tot = NB * cout;
        dense_kernel<<<(tot + 255) / 256, 256, 0, stream>>>(
            x, w, wstride, bias,
            bn ? bn->g : nullptr, bn ? bn->b : nullptr,
            bn ? bn->m : nullptr, bn ? bn->v : nullptr,
            y, cin, cout, relu, identk, tot);
    };

#define CONV(TIN, TOUT, RELU, RED, X, Y, W, BIAS, GMAX, CIN, CINP, COUT, WBS, BBS)      \
    conv1x1_wmma<TIN, TOUT, RELU, RED>                                                  \
        <<<dim3(NPTS / 64, (COUT) / 64, NB), 256, 0, stream>>>(                         \
            (const TIN*)(X), (TOUT*)(Y), (const f16*)(W), (const float*)(BIAS),         \
            (float*)(GMAX), CIN, CINP, COUT, NPTS, WBS, BBS)

    const int TOT_BN = NB * NPTS;

    auto run_cloud = [&](const float* ep, float* outF) {
        transpose_in<<<(TOT_BN + 255) / 256, 256, 0, stream>>>(ep, xT, NPTS, TOT_BN);
        // ---- STN3 ----
        CONV(float, f16, true,  false, xT, s1, pw_s1, pb_s1, nullptr, 3,   64,  64,   0, 0);
        CONV(f16,   f16, true,  false, s1, s2, pw_s2, pb_s2, nullptr, 64,  64,  128,  0, 0);
        fill_f32<<<(NB * 1024 + 255) / 256, 256, 0, stream>>>(pool, -INFINITY, NB * 1024);
        CONV(f16,   f16, true,  true,  s2, nullptr, pw_s3, pb_s3, pool, 128, 128, 1024, 0, 0);
        dense(pool, stn.f1.w, 1024, stn.f1.b, &stn.bn4, v512, 1024, 512, 1, 0);
        dense(v512, stn.f2.w, 512,  stn.f2.b, &stn.bn5, v256, 512,  256, 1, 0);
        dense(v256, stn.f3.w, 256,  stn.f3.b, nullptr,  tr3,  256,  9,   0, 3);
        apply_trans3<<<(TOT_BN + 255) / 256, 256, 0, stream>>>(xT, tr3, xe, NPTS, TOT_BN);
        // ---- encoder conv1 ----
        CONV(float, f16, true,  false, xe, e1, pw_e1, pb_e1, nullptr, 3,   64,  64,   0, 0);
        // ---- FSTN (k=64) ----
        CONV(f16,   f16, true,  false, e1, t1, pw_f1, pb_f1, nullptr, 64,  64,  64,   0, 0);
        CONV(f16,   f16, true,  false, t1, t2, pw_f2, pb_f2, nullptr, 64,  64,  128,  0, 0);
        fill_f32<<<(NB * 1024 + 255) / 256, 256, 0, stream>>>(pool, -INFINITY, NB * 1024);
        CONV(f16,   f16, true,  true,  t2, nullptr, pw_f3, pb_f3, pool, 128, 128, 1024, 0, 0);
        dense(pool, fstn.f1.w, 1024, fstn.f1.b, &fstn.bn4, v512, 1024, 512,  1, 0);
        dense(v512, fstn.f2.w, 512,  fstn.f2.b, &fstn.bn5, v256, 512,  256,  1, 0);
        dense(v256, fstn.f3.w, 256,  fstn.f3.b, nullptr,   tr64, 256,  4096, 0, 64);
        prep_transW<<<(NB * 4096 + 255) / 256, 256, 0, stream>>>(tr64, tw, 64, NB * 4096);
        // feature transform: per-batch 64x64 matmul over points
        CONV(f16,   f16, false, false, e1, pf, tw, nullptr, nullptr, 64, 64, 64, 4096, 0);
        // ---- encoder conv2/conv3(+fused global max) ----
        CONV(f16,   f16, true,  false, pf, e2, pw_e2, pb_e2, nullptr, 64,  64,  128,  0, 0);
        fill_f32<<<(NB * 1024 + 255) / 256, 256, 0, stream>>>(gmax, -INFINITY, NB * 1024);
        CONV(f16,   f16, false, true,  e2, nullptr, pw_e3, pb_e3, gmax, 128, 128, 1024, 0, 0);
        // head c1 split: per-batch bias = bn(Wg @ g + b)
        dense(gmax, nc1.w, 1088, nc1.b, &nbn1, gb, 1024, 512, 0, 0);
        CONV(f16,   f16, true,  false, pf, d1, pw_h1, gb,    nullptr, 64,  64,  512,  0, 512);
        CONV(f16,   f16, true,  false, d1, d2, pw_h2, pb_h2, nullptr, 512, 512, 256,  0, 0);
        CONV(f16,   f16, true,  false, d2, d3, pw_h3, pb_h3, nullptr, 256, 256, 128,  0, 0);
        CONV(f16,   float, true, false, d3, outF, pw_h4, pb_h4, nullptr, 128, 128, 64, 0, 0);
    };

    run_cloud(ep0, outA);
    run_cloud(ep1, outB);

    // ---- NFD + final head ----
    const int TOT_F = NB * 64 * NPTS;
    nfd_kernel<<<(TOT_F + 255) / 256, 256, 0, stream>>>(outA, outB, idx01, fd0, NPTS, TOT_F);
    nfd_kernel<<<(TOT_F + 255) / 256, 256, 0, stream>>>(outB, outA, idx10, fd1, NPTS, TOT_F);

    float* out = (float*)d_out;
    head_kernel<<<(TOT_BN + 255) / 256, 256, 0, stream>>>(
        fd0, mlp1.w, mlp1.b, bn_m1.g, bn_m1.b, bn_m1.m, bn_m1.v, mlp2w, out, NPTS);
    head_kernel<<<(TOT_BN + 255) / 256, 256, 0, stream>>>(
        fd1, mlp1.w, mlp1.b, bn_m1.g, bn_m1.b, bn_m1.m, bn_m1.v, mlp2w, out + TOT_BN, NPTS);
}
